// AggregationMPNN_78872779423809
// MI455X (gfx1250) — matrix operand order repaired and checked by reading, not compiled, not run on hardware
//
#include <hip/hip_runtime.h>
#include <hip/hip_bf16.h>

typedef __attribute__((ext_vector_type(16))) _Float16 v16h;
typedef __attribute__((ext_vector_type(8)))  float    v8f;
typedef __attribute__((ext_vector_type(4)))  float    v4f;

#define B_   512
#define N_   40
#define H_   128
#define FE_  16
#define M_   128
#define OUT_ 128
#define R_   (B_ * N_)   // 20480 rows

#define STRH 136   // LDS f16 row stride (272B: 16B aligned, dword-bank step 4)
#define STRC 264   // LDS f16 row stride for 256-wide cat tile (528B)
#define NPSTR 132  // LDS f32 row stride for NP/EM (528B: 16B aligned)

__device__ inline v8f wmma16(v16h a, v16h b, v8f c) {
  return __builtin_amdgcn_wmma_f32_16x16x32_f16(false, a, false, b, (short)0, c,
                                                false, false);
}
__device__ inline v8f v8zero() {
  v8f z; for (int i = 0; i < 8; ++i) z[i] = 0.f; return z;
}
__device__ inline v16h v16zero() {
  v16h z; for (int t = 0; t < 16; ++t) z[t] = (_Float16)0.f; return z;
}
__device__ inline float sigm(float x) { return 1.f / (1.f + __expf(-x)); }

// ---- CDNA5 async global->LDS DMA (ASYNCcnt-tracked), 16B per lane ----
__device__ inline unsigned lds_off(const void* p) {
  return (unsigned)(size_t)p;  // low 32 bits of generic LDS pointer = LDS byte addr
}
__device__ inline void async_copy_b128(unsigned lds_addr, const void* gptr) {
  asm volatile("global_load_async_to_lds_b128 %0, %1, off"
               :: "v"(lds_addr), "v"((unsigned long long)(size_t)gptr)
               : "memory");
}
__device__ inline void async_wait0() {
  asm volatile("s_wait_asynccnt 0x0" ::: "memory");
}

// A fragment (16x32 f16) from LDS row base (base = tile + (lane&15)*stride)
__device__ inline v16h load_afrag(const _Float16* base, int ks, int lane) {
  const _Float16* p = base + ks * 32 + ((lane & 16) ? 8 : 0);
  v16h a;
  ((v4f*)&a)[0] = *(const v4f*)p;        // K = k0..k0+7
  ((v4f*)&a)[1] = *(const v4f*)(p + 16); // K = k0+16..k0+23
  return a;
}
// B fragment (32x16 f16) from K-major transposed weights: WT[n*K + k]
__device__ inline v16h load_bfragT(const _Float16* WT, int K, int col, int ks, int lane) {
  const _Float16* p = WT + (size_t)col * K + ks * 32 + ((lane & 16) ? 8 : 0);
  v16h b;
  ((v4f*)&b)[0] = *(const v4f*)p;
  ((v4f*)&b)[1] = *(const v4f*)(p + 16);
  return b;
}

// ---------------- masks ----------------
__global__ void nghb_mask_kernel(const float* __restrict__ edges,
                                 unsigned char* __restrict__ nmask, int total) {
  int t = blockIdx.x * blockDim.x + threadIdx.x;
  if (t >= total) return;
  const float* p = edges + (size_t)t * FE_;
  float s = 0.f;
  for (int f = 0; f < FE_; ++f) s += p[f];
  nmask[t] = (s > 0.f) ? 1 : 0;
}
__global__ void node_mask_kernel(const unsigned char* __restrict__ nmask,
                                 unsigned char* __restrict__ nodeMask, int total) {
  int t = blockIdx.x * blockDim.x + threadIdx.x;
  if (t >= total) return;
  const unsigned char* p = nmask + (size_t)t * N_;
  unsigned char any = 0;
  for (int j = 0; j < N_; ++j) any |= p[j];
  nodeMask[t] = any ? 1 : 0;
}

// ---------------- once: f16 mirrors ----------------
__global__ void transpose_f16_kernel(const float* __restrict__ W,
                                     _Float16* __restrict__ WT, int K, int N) {
  int idx = blockIdx.x * blockDim.x + threadIdx.x;
  if (idx >= K * N) return;
  int k = idx / N, n = idx % N;
  WT[(size_t)n * K + k] = (_Float16)W[idx];
}
__global__ void cvt_nodes_kernel(const float* __restrict__ nodes,
                                 _Float16* __restrict__ nodesF,
                                 _Float16* __restrict__ hF, int total) {
  int t = blockIdx.x * blockDim.x + threadIdx.x;
  if (t >= total) return;
  float v = nodes[t];
  nodesF[t] = (_Float16)v;
  hF[t] = (_Float16)v;
}

// ---------------- NP = h@Wn, EM = h@Wm ----------------
__global__ __launch_bounds__(256) void proj_kernel(
    const _Float16* __restrict__ hF, const _Float16* __restrict__ WnT,
    const _Float16* __restrict__ WmT, float* __restrict__ NP,
    float* __restrict__ EM) {
  __shared__ __align__(16) _Float16 sA[64 * STRH];
  int tid = threadIdx.x;
  int rowBase = blockIdx.x * 64;
  // async DMA stage: 64 rows x 16 chunks of 8 halves
  for (int idx = tid; idx < 64 * 16; idx += 256) {
    int r = idx >> 4, c = idx & 15;
    async_copy_b128(lds_off(&sA[r * STRH + c * 8]),
                    hF + (size_t)(rowBase + r) * H_ + c * 8);
  }
  async_wait0();
  __syncthreads();
  int lane = tid & 31, wave = tid >> 5;
  // pairs of column tiles share one A fragment -> interleaved WMMA chains
  for (int pairId = wave; pairId < 32; pairId += 8) {
    int mat = pairId >> 4;          // 0 -> Wn/NP, 1 -> Wm/EM
    int rem = pairId & 15;
    int rt = rem >> 2, ctp = rem & 3;
    const _Float16* WT = mat ? WmT : WnT;
    float* OUTP = mat ? EM : NP;
    const _Float16* abase = sA + (rt * 16 + (lane & 15)) * STRH;
    int col0 = ctp * 32 + (lane & 15);
    int col1 = col0 + 16;
    v8f acc0 = v8zero(), acc1 = v8zero();
    for (int ks = 0; ks < 4; ++ks) {
      v16h a = load_afrag(abase, ks, lane);
      v16h b0 = load_bfragT(WT, 128, col0, ks, lane);
      v16h b1 = load_bfragT(WT, 128, col1, ks, lane);
      acc0 = wmma16(a, b0, acc0);
      acc1 = wmma16(a, b1, acc1);
    }
    int row0 = rowBase + rt * 16 + ((lane & 16) ? 8 : 0);
    for (int r = 0; r < 8; ++r) {
      OUTP[(size_t)(row0 + r) * M_ + col0] = acc0[r];
      OUTP[(size_t)(row0 + r) * M_ + col1] = acc1[r];
    }
  }
}

// ---------------- attention + messages (one block per graph b) ----------------
__global__ __launch_bounds__(512) void attn_kernel(
    const float* __restrict__ edges, const _Float16* __restrict__ WeT,
    const float* __restrict__ NP, const float* __restrict__ EM,
    const unsigned char* __restrict__ nmask, _Float16* __restrict__ MSGF) {
  __shared__ __align__(16) float sNP[N_ * NPSTR];
  __shared__ __align__(16) float sEM[N_ * NPSTR];
  __shared__ __align__(16) unsigned char sMask[N_ * N_];
  int b = blockIdx.x;
  int tid = threadIdx.x;
  // async DMA stage: NP/EM 40 rows x 32 chunks of 4 floats; mask 100 chunks
  for (int idx = tid; idx < N_ * 32; idx += 512) {
    int j = idx >> 5, c = idx & 31;
    const float* src = NP + (size_t)b * N_ * M_ + j * M_ + c * 4;
    async_copy_b128(lds_off(&sNP[j * NPSTR + c * 4]), src);
    const float* src2 = EM + (size_t)b * N_ * M_ + j * M_ + c * 4;
    async_copy_b128(lds_off(&sEM[j * NPSTR + c * 4]), src2);
  }
  for (int idx = tid; idx < 100; idx += 512)
    async_copy_b128(lds_off(&sMask[idx * 16]),
                    nmask + (size_t)b * N_ * N_ + idx * 16);
  async_wait0();
  __syncthreads();

  int lane = tid & 31, wave = tid >> 5;
  int rOff = (lane & 16) ? 8 : 0;
  for (int i = wave; i < N_; i += 16) {
    // prefetch next node's edge rows into cache
    if (i + 16 < N_) {
      const float* pf =
          edges + (((size_t)b * N_ + (i + 16)) * N_ + (lane & 15)) * FE_;
      __builtin_prefetch(pf, 0, 0);
    }
    // per-lane mask bits for the 24 (jt,r) slots this lane owns
    unsigned mbits = 0;
    for (int jt = 0; jt < 3; ++jt)
      for (int r = 0; r < 8; ++r) {
        int j = jt * 16 + r + rOff;
        if (j < N_ && sMask[i * N_ + j]) mbits |= 1u << (jt * 8 + r);
      }
    // A fragments: edges[b,i,j,0..15], K padded 16->32 with zeros
    v16h a[3];
    for (int jt = 0; jt < 3; ++jt) {
      v16h av = v16zero();
      int j = jt * 16 + (lane & 15);
      if (j < N_) {
        const float* rp =
            edges + (((size_t)b * N_ + i) * N_ + j) * FE_ + ((lane & 16) ? 8 : 0);
        v4f f0 = *(const v4f*)rp;
        v4f f1 = *(const v4f*)(rp + 4);
        av[0] = (_Float16)f0.x; av[1] = (_Float16)f0.y;
        av[2] = (_Float16)f0.z; av[3] = (_Float16)f0.w;
        av[4] = (_Float16)f1.x; av[5] = (_Float16)f1.y;
        av[6] = (_Float16)f1.z; av[7] = (_Float16)f1.w;
      }
      a[jt] = av;
    }
    for (int mt = 0; mt < 8; ++mt) {
      // B fragment from WeT (K=16 real; elems 0-7 hold K=8h..8h+7, elems 8-15 (K>=16) zero)
      v16h bf = v16zero();
      {
        const _Float16* p = WeT + (size_t)(mt * 16 + (lane & 15)) * FE_ + ((lane & 16) ? 8 : 0);
        ((v4f*)&bf)[0] = *(const v4f*)p;
      }
      int m = mt * 16 + (lane & 15);
      float num = 0.f, den = 0.f;
      for (int jt = 0; jt < 3; ++jt) {
        v8f c = v8zero();
        c = wmma16(a[jt], bf, c);   // ep tile: rows j, cols m
        for (int r = 0; r < 8; ++r) {
          if ((mbits >> (jt * 8 + r)) & 1) {
            int j = jt * 16 + r + rOff;
            float ev = tanhf(c[r] + sNP[j * NPSTR + m]);
            float p = __expf(ev);   // tanh in [-1,1]: no max-shift needed
            num += p * sEM[j * NPSTR + m];
            den += p;
          }
        }
      }
      num += __shfl_xor(num, 16, 32);
      den += __shfl_xor(den, 16, 32);
      if (lane < 16) {
        float msg = (den > 0.f) ? num / den : 0.f;
        MSGF[((size_t)b * N_ + i) * M_ + mt * 16 + lane] = (_Float16)msg;
      }
    }
  }
}

// ---------------- fused GRU update (h and hF updated in place) ----------------
__global__ __launch_bounds__(256) void gru_kernel(
    const _Float16* __restrict__ MSGF, float* __restrict__ h,
    _Float16* __restrict__ hF, const _Float16* __restrict__ WiT,
    const _Float16* __restrict__ WhT, const float* __restrict__ bi,
    const float* __restrict__ bh, const unsigned char* __restrict__ nodeMask) {
  __shared__ __align__(16) _Float16 sM[16 * STRH];
  __shared__ __align__(16) _Float16 sH[16 * STRH];
  int tid = threadIdx.x;
  int rowBase = blockIdx.x * 16;
  // async DMA stage: 16 rows x 16 chunks, both tiles
  for (int idx = tid; idx < 256; idx += 256) {
    int r = idx >> 4, c = idx & 15;
    async_copy_b128(lds_off(&sM[r * STRH + c * 8]),
                    MSGF + (size_t)(rowBase + r) * M_ + c * 8);
    async_copy_b128(lds_off(&sH[r * STRH + c * 8]),
                    hF + (size_t)(rowBase + r) * H_ + c * 8);
  }
  async_wait0();
  __syncthreads();
  int lane = tid & 31, wave = tid >> 5;
  v8f gi[3], gh[3];
  for (int p = 0; p < 3; ++p) { gi[p] = v8zero(); gh[p] = v8zero(); }
  const _Float16* amb = sM + (lane & 15) * STRH;
  const _Float16* ahb = sH + (lane & 15) * STRH;
  for (int ks = 0; ks < 4; ++ks) {
    v16h aM = load_afrag(amb, ks, lane);
    v16h aH = load_afrag(ahb, ks, lane);
    for (int p = 0; p < 3; ++p) {
      int col = p * 128 + wave * 16 + (lane & 15);
      v16h bI = load_bfragT(WiT, 128, col, ks, lane);
      v16h bH = load_bfragT(WhT, 128, col, ks, lane);
      gi[p] = wmma16(aM, bI, gi[p]);
      gh[p] = wmma16(aH, bH, gh[p]);
    }
  }
  int cc = wave * 16 + (lane & 15);
  float biR = bi[cc], biZ = bi[128 + cc], biN = bi[256 + cc];
  float bhR = bh[cc], bhZ = bh[128 + cc], bhN = bh[256 + cc];
  int rOff = (lane & 16) ? 8 : 0;
  for (int r = 0; r < 8; ++r) {
    int row = rowBase + r + rOff;
    float rg = sigm(gi[0][r] + biR + gh[0][r] + bhR);
    float zg = sigm(gi[1][r] + biZ + gh[1][r] + bhZ);
    float ng = tanhf(gi[2][r] + biN + rg * (gh[2][r] + bhN));
    float hold = h[(size_t)row * H_ + cc];
    float hnew = (1.f - zg) * ng + zg * hold;
    float hsel = nodeMask[row] ? hnew : hold;
    h[(size_t)row * H_ + cc] = hsel;
    hF[(size_t)row * H_ + cc] = (_Float16)hsel;
  }
}

// ---------------- gated readout ----------------
__global__ void zero_out_kernel(float* __restrict__ out, int total) {
  int t = blockIdx.x * blockDim.x + threadIdx.x;
  if (t < total) out[t] = 0.f;
}
__global__ __launch_bounds__(256) void readout_kernel(
    const _Float16* __restrict__ hF, const _Float16* __restrict__ nodesF,
    const _Float16* __restrict__ WgT, const float* __restrict__ bg,
    const _Float16* __restrict__ WoT, const float* __restrict__ bo,
    const unsigned char* __restrict__ nodeMask, float* __restrict__ out) {
  __shared__ __align__(16) _Float16 sC[16 * STRC];
  int tid = threadIdx.x;
  int rowBase = blockIdx.x * 16;
  // async DMA stage: cat = [h | nodes], 16 rows x 32 chunks
  for (int idx = tid; idx < 512; idx += 256) {
    int r = idx >> 5, c = idx & 31;
    const _Float16* src =
        (c < 16) ? hF + (size_t)(rowBase + r) * H_ + c * 8
                 : nodesF + (size_t)(rowBase + r) * H_ + (c - 16) * 8;
    async_copy_b128(lds_off(&sC[r * STRC + c * 8]), src);
  }
  async_wait0();
  __syncthreads();
  int lane = tid & 31, wave = tid >> 5;
  const _Float16* abase = sC + (lane & 15) * STRC;
  int col = wave * 16 + (lane & 15);
  v8f ag = v8zero(), ao = v8zero();
  for (int ks = 0; ks < 8; ++ks) {  // K = 256
    v16h a = load_afrag(abase, ks, lane);
    v16h bG = load_bfragT(WgT, 256, col, ks, lane);
    v16h bO = load_bfragT(WoT, 256, col, ks, lane);
    ag = wmma16(a, bG, ag);
    ao = wmma16(a, bO, ao);
  }
  float bgv = bg[col], bov = bo[col];
  int rOff = (lane & 16) ? 8 : 0;
  for (int r = 0; r < 8; ++r) {
    int row = rowBase + r + rOff;
    if (nodeMask[row]) {
      float g = sigm(ag[r] + bgv);
      float e = ao[r] + bov;
      atomicAdd(&out[(size_t)(row / N_) * OUT_ + col], g * e);
    }
  }
}

extern "C" void kernel_launch(void* const* d_in, const int* in_sizes, int n_in,
                              void* d_out, int out_size, void* d_ws, size_t ws_size,
                              hipStream_t stream) {
  (void)in_sizes; (void)n_in; (void)ws_size; (void)out_size;
  const float* nodes = (const float*)d_in[0];
  const float* edges = (const float*)d_in[1];
  const float* We = (const float*)d_in[2];
  const float* Wn = (const float*)d_in[3];
  const float* Wm = (const float*)d_in[4];
  const float* Wi = (const float*)d_in[5];
  const float* Wh = (const float*)d_in[6];
  const float* bi = (const float*)d_in[7];
  const float* bh = (const float*)d_in[8];
  const float* Wg = (const float*)d_in[9];
  const float* bg = (const float*)d_in[10];
  const float* Wo = (const float*)d_in[11];
  const float* bo = (const float*)d_in[12];
  float* out = (float*)d_out;

  char* ws = (char*)d_ws;
  size_t off = 0;
  float* h   = (float*)(ws + off); off += (size_t)R_ * H_ * 4;
  float* NP  = (float*)(ws + off); off += (size_t)R_ * M_ * 4;
  float* EM  = (float*)(ws + off); off += (size_t)R_ * M_ * 4;
  unsigned char* nmask    = (unsigned char*)(ws + off); off += (size_t)B_ * N_ * N_;
  unsigned char* nodeMask = (unsigned char*)(ws + off); off += (size_t)R_;
  _Float16* hF     = (_Float16*)(ws + off); off += (size_t)R_ * H_ * 2;
  _Float16* MSGF   = (_Float16*)(ws + off); off += (size_t)R_ * M_ * 2;
  _Float16* nodesF = (_Float16*)(ws + off); off += (size_t)R_ * H_ * 2;
  _Float16* WnT = (_Float16*)(ws + off); off += (size_t)128 * 128 * 2;
  _Float16* WmT = (_Float16*)(ws + off); off += (size_t)128 * 128 * 2;
  _Float16* WiT = (_Float16*)(ws + off); off += (size_t)128 * 384 * 2;
  _Float16* WhT = (_Float16*)(ws + off); off += (size_t)128 * 384 * 2;
  _Float16* WgT = (_Float16*)(ws + off); off += (size_t)256 * 128 * 2;
  _Float16* WoT = (_Float16*)(ws + off); off += (size_t)256 * 128 * 2;
  _Float16* WeT = (_Float16*)(ws + off); off += (size_t)16 * 128 * 2;

  // h = nodes (f32) ; hF = nodesF = f16(nodes)
  hipMemcpyAsync(h, nodes, (size_t)R_ * H_ * 4, hipMemcpyDeviceToDevice, stream);
  cvt_nodes_kernel<<<(R_ * H_ + 255) / 256, 256, 0, stream>>>(nodes, nodesF, hF,
                                                              R_ * H_);

  // masks
  nghb_mask_kernel<<<(B_ * N_ * N_ + 255) / 256, 256, 0, stream>>>(edges, nmask,
                                                                   B_ * N_ * N_);
  node_mask_kernel<<<(R_ + 255) / 256, 256, 0, stream>>>(nmask, nodeMask, R_);

  // f16 K-major weight transposes
  transpose_f16_kernel<<<(128 * 128 + 255) / 256, 256, 0, stream>>>(Wn, WnT, 128, 128);
  transpose_f16_kernel<<<(128 * 128 + 255) / 256, 256, 0, stream>>>(Wm, WmT, 128, 128);
  transpose_f16_kernel<<<(128 * 384 + 255) / 256, 256, 0, stream>>>(Wi, WiT, 128, 384);
  transpose_f16_kernel<<<(128 * 384 + 255) / 256, 256, 0, stream>>>(Wh, WhT, 128, 384);
  transpose_f16_kernel<<<(256 * 128 + 255) / 256, 256, 0, stream>>>(Wg, WgT, 256, 128);
  transpose_f16_kernel<<<(256 * 128 + 255) / 256, 256, 0, stream>>>(Wo, WoT, 256, 128);
  transpose_f16_kernel<<<(16 * 128 + 255) / 256, 256, 0, stream>>>(We, WeT, 16, 128);

  for (int pass = 0; pass < 3; ++pass) {
    proj_kernel<<<R_ / 64, 256, 0, stream>>>(hF, WnT, WmT, NP, EM);
    attn_kernel<<<B_, 512, 0, stream>>>(edges, WeT, NP, EM, nmask, MSGF);
    gru_kernel<<<R_ / 16, 256, 0, stream>>>(MSGF, h, hF, WiT, WhT, bi, bh,
                                            nodeMask);
  }

  zero_out_kernel<<<(B_ * OUT_ + 255) / 256, 256, 0, stream>>>(out, B_ * OUT_);
  readout_kernel<<<R_ / 16, 256, 0, stream>>>(hF, nodesF, WgT, bg, WoT, bo,
                                              nodeMask, out);
}
